// _ALSTM_51745765983011
// MI455X (gfx1250) — compile-verified, matrix operand order checked by reading, not run on hardware
//
#include <hip/hip_runtime.h>
#include <hip/hip_bf16.h>
#include <math.h>

// ---------------------------------------------------------------------------
// ALSTM for MI455X (gfx1250).  B=32, T=256, D=512, H=512, A=128, 2 directions.
// Strategy:
//   1) cvt x -> bf16
//   2) pack all weights into WMMA B-fragment layout (bf16)
//   3) big WMMA GEMMs: Xpre = x@Wx+b (fragment-major), aXpre = x@aWx+ab
//   4) persistent recurrent kernel: 1 workgroup (32 waves) per direction,
//      h/ah state in LDS (bf16), c/ac state in per-wave VGPR fragments,
//      all matmuls via v_wmma_f32_16x16x32_bf16.
// ---------------------------------------------------------------------------

typedef __attribute__((ext_vector_type(16))) __bf16 v16bf;
typedef __attribute__((ext_vector_type(8)))  __bf16 v8bf;
typedef __attribute__((ext_vector_type(8)))  float  v8f;

#define Bn 32
#define Tn 256
#define Dn 512
#define Hn 512
#define An 128

__device__ __forceinline__ float sigf(float x) { return 1.0f / (1.0f + __expf(-x)); }

// Load a 16x32 bf16 A-fragment for one lane.
// Per ISA 7.12.2 (16-bit A 16x32): lanes 0-15 hold M=0..15, K={0..7,16..23};
// lanes 16-31 hold M=0..15, K={8..15,24..31}.  `p` must already point at
// row (M = lane&15), column (k0 + (lane>>4)*8).  We then grab 8 contiguous
// elements and 8 more at +16.
__device__ __forceinline__ v16bf load_afrag(const __bf16* p) {
    union { v16bf v; v8bf h[2]; } u;
    u.h[0] = *(const v8bf*)(p);
    u.h[1] = *(const v8bf*)(p + 16);
    return u.v;
}

// ---------------------------------------------------------------------------
// fp32 -> bf16 elementwise convert
// ---------------------------------------------------------------------------
__global__ __launch_bounds__(256) void cvt_bf16_kernel(const float* __restrict__ x,
                                                       __bf16* __restrict__ y, int n) {
    int i = blockIdx.x * 256 + threadIdx.x;
    if (i < n) y[i] = (__bf16)x[i];
}

// ---------------------------------------------------------------------------
// Pack fp32 weight W (K x N, row-major) into bf16 B-fragment tiles.
// Tile (nt, kt) covers cols [nt*16, nt*16+16), rows [kt*32, kt*32+32).
// Packed element layout matches what the GEMM reads:
//   packed[((nt*KT + kt)*32 + lane)*16 + j]
//   lane&15 -> column, lane>>4 -> K-half, j -> K offset within half pattern.
// ---------------------------------------------------------------------------
__global__ __launch_bounds__(256) void pack_w_kernel(const float* __restrict__ W,
                                                     __bf16* __restrict__ Wp,
                                                     int K, int N) {
    int idx = blockIdx.x * 256 + threadIdx.x;
    int total = K * N;
    if (idx >= total) return;
    int KT = K >> 5;
    int tile = idx >> 9;         // / 512
    int e    = idx & 511;
    int lane = e >> 4;
    int j    = e & 15;
    int nt = tile / KT;
    int kt = tile - nt * KT;
    int n  = nt * 16 + (lane & 15);
    int kh = lane >> 4;
    int koff = (j < 8) ? (kh * 8 + j) : (16 + kh * 8 + (j - 8));
    int k = kt * 32 + koff;
    Wp[idx] = (__bf16)W[(size_t)k * N + n];
}

// ---------------------------------------------------------------------------
// Pre-GEMM: pre = x @ W + bias, output in fragment-major layout.
// A = xbf, laid out (B, T, D) bf16.  Rows of a tile are batch indices.
// One wave per output 16x16 tile; tile id gw = (t*NT + nt)*2 + rowblk.
// K = D = 512 -> 16 WMMA steps.  Grid is sized exactly (tiles % 8 == 0),
// so no bounds check -> EXEC stays all-ones for WMMA.
// ---------------------------------------------------------------------------
__global__ __launch_bounds__(256) void gemm_pre_kernel(const __bf16* __restrict__ xbf,
                                                       const __bf16* __restrict__ Wp,
                                                       const float* __restrict__ bias,
                                                       float* __restrict__ pre,
                                                       int NT) {
    int lane = threadIdx.x & 31;
    int wlocal = __builtin_amdgcn_readfirstlane((int)(threadIdx.x >> 5));
    int gw = blockIdx.x * 8 + wlocal;

    int t   = gw / (NT * 2);
    int rem = gw - t * (NT * 2);
    int nt  = rem >> 1;
    int rb  = rem & 1;

    float bv = bias[nt * 16 + (lane & 15)];
    v8f acc = {bv, bv, bv, bv, bv, bv, bv, bv};

    int m  = rb * 16 + (lane & 15);           // batch row
    int kh = (lane >> 4) & 1;
    const __bf16* arow = xbf + ((size_t)m * Tn + t) * Dn + kh * 8;
    const __bf16* bp   = Wp + ((size_t)nt * 16) * 512 + (size_t)lane * 16;

#pragma unroll
    for (int kt = 0; kt < 16; ++kt) {
        v16bf af = load_afrag(arow + kt * 32);
        v16bf bf = *(const v16bf*)(bp + (size_t)kt * 512);
        acc = __builtin_amdgcn_wmma_f32_16x16x32_bf16(false, af, false, bf,
                                                      (short)0, acc, false, false);
    }
    *(v8f*)(pre + ((size_t)gw * 32 + lane) * 8) = acc;
}

// ---------------------------------------------------------------------------
// Recurrent kernel: grid.x = 2 (direction), 1024 threads = 32 waves.
// LDS: h (32x512 bf16), ah (32x128 bf16).  c / ac live in VGPR fragments.
// ---------------------------------------------------------------------------
__global__ __launch_bounds__(1024) void alstm_recur_kernel(
        const float* __restrict__ preX_base, const float* __restrict__ preA_base,
        const __bf16* __restrict__ Whp_base, const __bf16* __restrict__ Wzp_base,
        const __bf16* __restrict__ aWhp_base,
        float* __restrict__ out,
        size_t sPreX, size_t sPreA, size_t sWh, size_t sWz, size_t saWh) {
    int dir = blockIdx.x;
    const float*  preX = preX_base + (size_t)dir * sPreX;
    const float*  preA = preA_base + (size_t)dir * sPreA;
    const __bf16* Whp  = Whp_base  + (size_t)dir * sWh;
    const __bf16* Wzp  = Wzp_base  + (size_t)dir * sWz;
    const __bf16* aWhp = aWhp_base + (size_t)dir * saWh;

    int tid  = threadIdx.x;
    int lane = tid & 31;
    int w    = __builtin_amdgcn_readfirstlane(tid >> 5);   // uniform wave id

    __shared__ __align__(32) __bf16 h_lds[Bn * Hn];    // 32 KB
    __shared__ __align__(32) __bf16 ah_lds[Bn * An];   // 8 KB

    for (int i = tid; i < Bn * Hn; i += 1024) h_lds[i]  = (__bf16)0.0f;
    for (int i = tid; i < Bn * An; i += 1024) ah_lds[i] = (__bf16)0.0f;

    v8f c_frag0 = {}, c_frag1 = {};   // main-cell c, one 16x16 tile per rowblk
    v8f ac_frag = {};                 // adaptive-cell c (waves 0..15 only)

    // assignments
    int a_rb = (w >> 3) & 1;          // adaptive rowblk (waves 0..15)
    int a_nt = w & 7;                 // adaptive col group (0..7 of A=128)
    int m_nt = w;                     // main col group (0..31 of H=512)
    int mrow = lane & 15;
    int kh   = (lane >> 4) & 1;
    bool aw  = (w < 16);

    __syncthreads();

#pragma unroll 1
    for (int t = 0; t < Tn; ++t) {
        // ---------------- adaptive cell ----------------
        v8f ahv = {};
        if (aw) {
            v8f acc[4];
#pragma unroll
            for (int g = 0; g < 4; ++g) {
                size_t tile = ((size_t)t * 32 + (size_t)(g * 8 + a_nt)) * 2 + a_rb;
                acc[g] = *(const v8f*)(preA + (tile * 32 + lane) * 8);
            }
            const __bf16* arow = ah_lds + (size_t)(a_rb * 16 + mrow) * An + kh * 8;
#pragma unroll
            for (int kt = 0; kt < 4; ++kt) {
                v16bf af = load_afrag(arow + kt * 32);
#pragma unroll
                for (int g = 0; g < 4; ++g) {
                    int ntg = g * 8 + a_nt;
                    v16bf bf = *(const v16bf*)(aWhp + (((size_t)ntg * 4 + kt) * 32 + lane) * 16);
                    acc[g] = __builtin_amdgcn_wmma_f32_16x16x32_bf16(false, af, false, bf,
                                                                     (short)0, acc[g], false, false);
                }
            }
#pragma unroll
            for (int r = 0; r < 8; ++r) {
                float gi = acc[0][r], gf = acc[1][r], gu = acc[2][r], go = acc[3][r];
                float acv = sigf(gf) * ac_frag[r] + sigf(gi) * tanhf(gu);
                ac_frag[r] = acv;
                ahv[r] = sigf(go) * tanhf(acv);
            }
        }
        __syncthreads();   // all reads of old ah done
        if (aw) {
#pragma unroll
            for (int r = 0; r < 8; ++r) {
                int m = a_rb * 16 + r + 8 * kh;
                int n = a_nt * 16 + (lane & 15);
                ah_lds[(size_t)m * An + n] = (__bf16)ahv[r];
            }
        }
        __syncthreads();   // new ah visible

        // ---------------- main cell ----------------
        v8f hv[2];
#pragma unroll
        for (int s = 0; s < 2; ++s) {
            int rb = s;
            v8f acc[4];
#pragma unroll
            for (int g = 0; g < 4; ++g) {
                size_t tile = ((size_t)t * 128 + (size_t)(g * 32 + m_nt)) * 2 + rb;
                acc[g] = *(const v8f*)(preX + (tile * 32 + lane) * 8);
            }
            // h_{t-1} @ Wh : K = 512
            const __bf16* hrow = h_lds + (size_t)(rb * 16 + mrow) * Hn + kh * 8;
#pragma unroll
            for (int kt = 0; kt < 16; ++kt) {
                v16bf af = load_afrag(hrow + kt * 32);
#pragma unroll
                for (int g = 0; g < 4; ++g) {
                    int ntg = g * 32 + m_nt;
                    v16bf bf = *(const v16bf*)(Whp + (((size_t)ntg * 16 + kt) * 32 + lane) * 16);
                    acc[g] = __builtin_amdgcn_wmma_f32_16x16x32_bf16(false, af, false, bf,
                                                                     (short)0, acc[g], false, false);
                }
            }
            // ah_t @ Wz : K = 128
            const __bf16* zrow = ah_lds + (size_t)(rb * 16 + mrow) * An + kh * 8;
#pragma unroll
            for (int kt = 0; kt < 4; ++kt) {
                v16bf af = load_afrag(zrow + kt * 32);
#pragma unroll
                for (int g = 0; g < 4; ++g) {
                    int ntg = g * 32 + m_nt;
                    v16bf bf = *(const v16bf*)(Wzp + (((size_t)ntg * 4 + kt) * 32 + lane) * 16);
                    acc[g] = __builtin_amdgcn_wmma_f32_16x16x32_bf16(false, af, false, bf,
                                                                     (short)0, acc[g], false, false);
                }
            }
            v8f& cf = (s == 0) ? c_frag0 : c_frag1;
#pragma unroll
            for (int r = 0; r < 8; ++r) {
                float gi = acc[0][r], gf = acc[1][r], gu = acc[2][r], go = acc[3][r];
                float cv = sigf(gf) * cf[r] + sigf(gi) * tanhf(gu);
                cf[r] = cv;
                hv[s][r] = sigf(go) * tanhf(cv);
            }
        }
        __syncthreads();   // all reads of old h done
#pragma unroll
        for (int s = 0; s < 2; ++s) {
#pragma unroll
            for (int r = 0; r < 8; ++r) {
                int m = s * 16 + r + 8 * kh;          // batch
                int n = m_nt * 16 + (lane & 15);      // hidden index
                h_lds[(size_t)m * Hn + n] = (__bf16)hv[s][r];
                out[((size_t)m * Tn + t) * (2 * Hn) + (size_t)dir * Hn + n] = hv[s][r];
            }
        }
        __syncthreads();   // new h visible
    }
}

// ---------------------------------------------------------------------------
// Host-side orchestration
// ---------------------------------------------------------------------------
extern "C" void kernel_launch(void* const* d_in, const int* in_sizes, int n_in,
                              void* d_out, int out_size, void* d_ws, size_t ws_size,
                              hipStream_t stream) {
    (void)in_sizes; (void)n_in; (void)out_size; (void)ws_size;

    const float* x      = (const float*)d_in[0];
    const float* Wx[2]  = {(const float*)d_in[1],  (const float*)d_in[8]};
    const float* Wh[2]  = {(const float*)d_in[2],  (const float*)d_in[9]};
    const float* bb[2]  = {(const float*)d_in[3],  (const float*)d_in[10]};
    const float* aWx[2] = {(const float*)d_in[4],  (const float*)d_in[11]};
    const float* aWh[2] = {(const float*)d_in[5],  (const float*)d_in[12]};
    const float* ab[2]  = {(const float*)d_in[6],  (const float*)d_in[13]};
    const float* Wz[2]  = {(const float*)d_in[7],  (const float*)d_in[14]};

    // workspace carve-up (all offsets 256B aligned)
    char* ws = (char*)d_ws;
    size_t off = 0;
    auto carve = [&](size_t bytes) -> char* {
        char* p = ws + off;
        off += (bytes + 255) & ~(size_t)255;
        return p;
    };

    const size_t N_X    = (size_t)Bn * Tn * Dn;          // 4,194,304
    const size_t N_WXP  = (size_t)Dn * 4 * Hn;           // 1,048,576 per dir
    const size_t N_WHP  = (size_t)Hn * 4 * Hn;           // 1,048,576 per dir
    const size_t N_WZP  = (size_t)An * 4 * Hn;           //   262,144 per dir
    const size_t N_AWXP = (size_t)Dn * 4 * An;           //   262,144 per dir
    const size_t N_AWHP = (size_t)An * 4 * An;           //    65,536 per dir
    const size_t N_PREX = (size_t)Tn * 128 * 2 * 256;    // 16,777,216 f32 per dir
    const size_t N_PREA = (size_t)Tn * 32 * 2 * 256;     //  4,194,304 f32 per dir

    __bf16* xbf  = (__bf16*)carve(N_X * 2);
    __bf16* wxp  = (__bf16*)carve(2 * N_WXP * 2);
    __bf16* whp  = (__bf16*)carve(2 * N_WHP * 2);
    __bf16* wzp  = (__bf16*)carve(2 * N_WZP * 2);
    __bf16* awxp = (__bf16*)carve(2 * N_AWXP * 2);
    __bf16* awhp = (__bf16*)carve(2 * N_AWHP * 2);
    float*  preX = (float*)carve(2 * N_PREX * 4);
    float*  preA = (float*)carve(2 * N_PREA * 4);

    // 1) convert x to bf16
    cvt_bf16_kernel<<<(int)(N_X / 256), 256, 0, stream>>>(x, xbf, (int)N_X);

    // 2) pack weights + 3) precompute GEMMs, per direction
    for (int d = 0; d < 2; ++d) {
        pack_w_kernel<<<(int)(N_WXP  / 256), 256, 0, stream>>>(Wx[d],  wxp  + d * N_WXP,  Dn, 4 * Hn);
        pack_w_kernel<<<(int)(N_WHP  / 256), 256, 0, stream>>>(Wh[d],  whp  + d * N_WHP,  Hn, 4 * Hn);
        pack_w_kernel<<<(int)(N_WZP  / 256), 256, 0, stream>>>(Wz[d],  wzp  + d * N_WZP,  An, 4 * Hn);
        pack_w_kernel<<<(int)(N_AWXP / 256), 256, 0, stream>>>(aWx[d], awxp + d * N_AWXP, Dn, 4 * An);
        pack_w_kernel<<<(int)(N_AWHP / 256), 256, 0, stream>>>(aWh[d], awhp + d * N_AWHP, An, 4 * An);

        // Xpre: tiles = T*128*2 = 65536 -> 8192 blocks of 8 waves
        gemm_pre_kernel<<<8192, 256, 0, stream>>>(xbf, wxp + d * N_WXP, bb[d],
                                                  preX + d * N_PREX, 128);
        // aXpre: tiles = T*32*2 = 16384 -> 2048 blocks
        gemm_pre_kernel<<<2048, 256, 0, stream>>>(xbf, awxp + d * N_AWXP, ab[d],
                                                  preA + d * N_PREA, 32);
    }

    // 4) recurrent scan, one workgroup per direction
    alstm_recur_kernel<<<2, 1024, 0, stream>>>(preX, preA, whp, wzp, awhp,
                                               (float*)d_out,
                                               N_PREX, N_PREA, N_WHP, N_WZP, N_AWHP);
}
